// PrevTrajEncoder_64020782514430
// MI455X (gfx1250) — compile-verified, matrix operand order checked by reading, not run on hardware
//
#include <hip/hip_runtime.h>
#include <math.h>

// ---------------------------------------------------------------------------
// PrevTrajEncoder for MI455X (gfx1250, wave32).
// GEMMs: v_wmma_f32_16x16x32_bf16, 2x2 tiles per wave (4 WMMA / 8 frag loads).
// GRU scan: single workgroup (serial dependence), h in LDS, w_hh pre-packed as
// transposed bf16 pairs for coalesced L2 streaming; next-step gi prefetched
// via global_prefetch_b8.
// ---------------------------------------------------------------------------

typedef __attribute__((ext_vector_type(16))) __bf16 v16bf;
typedef __attribute__((ext_vector_type(8)))  float  v8f;

#define TSEQ   512
#define DPOI   256
#define DIN    260
#define DINP   288    // DIN padded to multiple of 32 (K for GEMM1)
#define DHID   356
#define DHIDP  384    // DHID padded (N of GEMM1, K of GEMM2)
#define DOUT   512
#define G3     1536   // 3*DOUT

// ---- bf16 helpers (bit-level; avoids scalar __bf16 ABI) --------------------
static __device__ __forceinline__ unsigned short f2bf(float x) {
  unsigned int u = __float_as_uint(x);
  if ((u & 0x7fffffffu) > 0x7f800000u) return (unsigned short)((u >> 16) | 0x0040u);
  unsigned int r = u + 0x7fffu + ((u >> 16) & 1u);   // round-to-nearest-even
  return (unsigned short)(r >> 16);
}
static __device__ __forceinline__ float bf_lo(unsigned int u) { return __uint_as_float(u << 16); }
static __device__ __forceinline__ float bf_hi(unsigned int u) { return __uint_as_float(u & 0xffff0000u); }

// Load a 16x32 bf16 WMMA fragment row for this lane.
// Per ISA 7.12.2: lane<16 -> K{0..7,16..23}, lane>=16 -> K{8..15,24..31};
// p must already point at row*ld + 8*(lane>>4). Two 16B loads.
static __device__ __forceinline__ v16bf ld_frag(const unsigned short* p) {
  union { uint4 q[2]; v16bf v; } u;
  u.q[0] = *(const uint4*)(p);
  u.q[1] = *(const uint4*)(p + 16);
  return u.v;
}

static __device__ __forceinline__ v8f wmma_bf16(const v16bf& a, const v16bf& b, const v8f& c) {
  return __builtin_amdgcn_wmma_f32_16x16x32_bf16(false, a, false, b, (short)0, c, false, false);
}

// ---- weight prep -----------------------------------------------------------
// W (rows=K x cols=N, f32 row-major)  ->  Wt (nt x ldt, bf16, N x Kpad, zero pad)
__global__ void prep_transpose(const float* __restrict__ W, int rows, int cols,
                               unsigned short* __restrict__ Wt, int ldt, int nt) {
  int i = blockIdx.x * blockDim.x + threadIdx.x;
  int total = nt * ldt;
  if (i >= total) return;
  int n = i / ldt, k = i - n * ldt;
  float v = (n < cols && k < rows) ? W[(size_t)k * cols + n] : 0.0f;
  Wt[i] = f2bf(v);
}

__global__ void prep_copy_bf16(const float* __restrict__ W,
                               unsigned short* __restrict__ Wb, int n) {
  int i = blockIdx.x * blockDim.x + threadIdx.x;
  if (i < n) Wb[i] = f2bf(W[i]);
}

// w_hh (1536 x 512 f32) -> whhP: packed bf16 pairs, transposed:
// whhP[m*1536 + c] = {w_hh[c][2m] , w_hh[c][2m+1]}  (m = 0..255, c = 0..1535)
// -> coalesced 4B/lane loads in the GRU scan inner loop.
__global__ void prep_whh_packed(const float* __restrict__ whh,
                                unsigned int* __restrict__ out) {
  int i = blockIdx.x * blockDim.x + threadIdx.x;
  if (i >= 256 * G3) return;
  int m = i / G3, c = i - m * G3;
  unsigned int lo = f2bf(whh[(size_t)c * DOUT + 2 * m]);
  unsigned int hi = f2bf(whh[(size_t)c * DOUT + 2 * m + 1]);
  out[i] = lo | (hi << 16);
}

// ---- stage 1: gather + features + LayerNorm -> x bf16 (512 x 288) ----------
__global__ void featurize(const int* __restrict__ idx, const float* __restrict__ timef,
                          const float* __restrict__ emb, const float* __restrict__ latlon,
                          const float* __restrict__ lnw, const float* __restrict__ lnb,
                          unsigned short* __restrict__ xb) {
  __shared__ float feat[4];
  __shared__ float red[256];
  int t = blockIdx.x;
  int i = threadIdx.x;                 // 0..255
  int id = idx[t];
  float e = emb[(size_t)id * DPOI + i];

  if (i == 0) {
    const float TWO_PI = 6.2831853071795864769f;
    float tc = fminf(fmaxf(timef[t], 0.0f), 1.0f);
    feat[0] = sinf(TWO_PI * tc);
    feat[1] = cosf(TWO_PI * tc);
    float dt_h = 0.0f, dd_km = 0.0f;
    if (t > 0) {
      float tp = fminf(fmaxf(timef[t - 1], 0.0f), 1.0f);
      float d = tc - tp;
      d = d - floorf(d);               // jnp.mod(.,1.0): non-negative
      dt_h = d * 24.0f;
      int idp = idx[t - 1];
      float la1 = fminf(fmaxf(latlon[(size_t)idp * 2 + 0], -90.f), 90.f)  * 0.017453292519943295f;
      float lo1 = fminf(fmaxf(latlon[(size_t)idp * 2 + 1], -180.f), 180.f) * 0.017453292519943295f;
      float la2 = fminf(fmaxf(latlon[(size_t)id  * 2 + 0], -90.f), 90.f)  * 0.017453292519943295f;
      float lo2 = fminf(fmaxf(latlon[(size_t)id  * 2 + 1], -180.f), 180.f) * 0.017453292519943295f;
      float sdl = sinf((la2 - la1) * 0.5f);
      float sdo = sinf((lo2 - lo1) * 0.5f);
      float c1 = fminf(fmaxf(cosf(la1), -1.f), 1.f);
      float c2 = fminf(fmaxf(cosf(la2), -1.f), 1.f);
      float a = sdl * sdl + c1 * c2 * sdo * sdo;
      float a0 = fminf(fmaxf(a, 0.f), 1.f);
      float a1 = fminf(fmaxf(1.0f - a, 1e-12f), 1.f);
      float c = 2.0f * atan2f(sqrtf(a0), sqrtf(a1));
      dd_km = 6371.0088f * c;
    }
    feat[2] = log1pf(dt_h);
    feat[3] = log1pf(dd_km);
  }
  __syncthreads();

  // mean over 260 elements
  float part = e + ((i < 4) ? feat[i] : 0.0f);
  red[i] = part; __syncthreads();
  for (int o = 128; o > 0; o >>= 1) { if (i < o) red[i] += red[i + o]; __syncthreads(); }
  float mu = red[0] / (float)DIN;
  __syncthreads();

  float dv = (e - mu) * (e - mu);
  if (i < 4) { float f = feat[i] - mu; dv += f * f; }
  red[i] = dv; __syncthreads();
  for (int o = 128; o > 0; o >>= 1) { if (i < o) red[i] += red[i + o]; __syncthreads(); }
  float rstd = rsqrtf(red[0] / (float)DIN + 1e-5f);

  unsigned short* row = xb + (size_t)t * DINP;
  row[i] = f2bf((e - mu) * rstd * lnw[i] + lnb[i]);
  if (i < 32) {                        // features at 256..259, zero pad 260..287
    int c2 = DPOI + i;
    float v = 0.0f;
    if (i < 4) v = (feat[i] - mu) * rstd * lnw[c2] + lnb[c2];
    row[c2] = f2bf(v);
  }
}

// ---- bf16 WMMA GEMM, 32x32 tile per wave (2x2 WMMA tiles) ------------------
// C[M x N] = A[M x K] * Bt[N x K]^T + bias, optional exact GELU.
// C/D layout: lane -> column, VGPR v -> row v + 8*(lane>=16).
static __device__ __forceinline__ void epilogue_tile(
    const v8f& c, int rowBase, int colBase, int l, int half,
    const float* __restrict__ bias, int nbias,
    float* __restrict__ outF, unsigned short* __restrict__ outB,
    int ldc, int act) {
  int n = colBase + l;
  float bv = (n < nbias) ? bias[n] : 0.0f;
#pragma unroll
  for (int v = 0; v < 8; ++v) {
    int row = rowBase + v + 8 * half;
    float val = c[v] + bv;
    if (act) val = 0.5f * val * (1.0f + erff(val * 0.70710678118654752f));
    size_t o = (size_t)row * ldc + n;
    if (outF) outF[o] = val;
    else      outB[o] = f2bf(val);
  }
}

__global__ void wmma_gemm(const unsigned short* __restrict__ A, int lda,
                          const unsigned short* __restrict__ Bt, int ldb,
                          const float* __restrict__ bias, int nbias,
                          int mt, int nt, int kdim,     // 32x32 tile counts
                          float* __restrict__ outF, unsigned short* __restrict__ outB,
                          int ldc, int act) {
  int wave = blockIdx.x * (blockDim.x >> 5) + (threadIdx.x >> 5);
  if (wave >= mt * nt) return;
  int lane = threadIdx.x & 31;
  int tm = wave / nt, tn = wave - tm * nt;
  int half = lane >> 4, l = lane & 15;
  const unsigned short* a0 = A  + (size_t)(tm * 32 + l) * lda + 8 * half;
  const unsigned short* a1 = a0 + (size_t)16 * lda;
  const unsigned short* b0 = Bt + (size_t)(tn * 32 + l) * ldb + 8 * half;
  const unsigned short* b1 = b0 + (size_t)16 * ldb;
  v8f c00 = {}, c01 = {}, c10 = {}, c11 = {};
  for (int k = 0; k < kdim; k += 32) {
    v16bf A0 = ld_frag(a0 + k);
    v16bf A1 = ld_frag(a1 + k);
    v16bf B0 = ld_frag(b0 + k);
    v16bf B1 = ld_frag(b1 + k);
    c00 = wmma_bf16(A0, B0, c00);
    c01 = wmma_bf16(A0, B1, c01);
    c10 = wmma_bf16(A1, B0, c10);
    c11 = wmma_bf16(A1, B1, c11);
  }
  int r0 = tm * 32, n0 = tn * 32;
  epilogue_tile(c00, r0,      n0,      l, half, bias, nbias, outF, outB, ldc, act);
  epilogue_tile(c01, r0,      n0 + 16, l, half, bias, nbias, outF, outB, ldc, act);
  epilogue_tile(c10, r0 + 16, n0,      l, half, bias, nbias, outF, outB, ldc, act);
  epilogue_tile(c11, r0 + 16, n0 + 16, l, half, bias, nbias, outF, outB, ldc, act);
}

// ---- stage 4: GRU scan (strictly sequential; single workgroup) -------------
__global__ void __launch_bounds__(512) gru_scan(const float* __restrict__ gi,
                                                const unsigned int* __restrict__ whhP,
                                                const float* __restrict__ b_hh,
                                                float* __restrict__ hs) {
  __shared__ float h[DOUT];
  int j = threadIdx.x;                 // 0..511
  h[j] = 0.0f;
  float hj = 0.0f;
  float bhr = b_hh[j], bhz = b_hh[DOUT + j], bhn = b_hh[2 * DOUT + j];
  const unsigned int* pr = whhP + j;
  __syncthreads();
  for (int t = 0; t < TSEQ; ++t) {
    // hide next step's gi fetch behind this step's dot products
    if (t + 1 < TSEQ) {
      const float* gn = gi + (size_t)(t + 1) * G3;
      __builtin_prefetch(&gn[j], 0, 3);            // global_prefetch_b8
      __builtin_prefetch(&gn[DOUT + j], 0, 3);
      __builtin_prefetch(&gn[2 * DOUT + j], 0, 3);
    }
    float dr = 0.f, dz = 0.f, dn = 0.f;
#pragma unroll 4
    for (int m = 0; m < 256; ++m) {
      float2 hp = *(const float2*)&h[2 * m];       // LDS broadcast pair
      unsigned int ur = pr[(size_t)m * G3];
      unsigned int uz = pr[(size_t)m * G3 + DOUT];
      unsigned int un = pr[(size_t)m * G3 + 2 * DOUT];
      dr += bf_lo(ur) * hp.x + bf_hi(ur) * hp.y;
      dz += bf_lo(uz) * hp.x + bf_hi(uz) * hp.y;
      dn += bf_lo(un) * hp.x + bf_hi(un) * hp.y;
    }
    const float* g = gi + (size_t)t * G3;          // b_ih already folded in
    float r = 1.0f / (1.0f + __expf(-(g[j] + dr + bhr)));
    float z = 1.0f / (1.0f + __expf(-(g[DOUT + j] + dz + bhz)));
    float n = tanhf(g[2 * DOUT + j] + r * (dn + bhn));
    float hn = (1.0f - z) * n + z * hj;
    __syncthreads();
    h[j] = hn; hj = hn;
    hs[(size_t)t * DOUT + j] = hn;
    __syncthreads();
  }
}

// ---- stage 5: attention pooling -> out[0:512]=summary, out[512:1024]=attn --
__global__ void __launch_bounds__(512) attn_summary(const float* __restrict__ hs,
                                                    const float* __restrict__ q,
                                                    const float* __restrict__ loglam,
                                                    float* __restrict__ out) {
  __shared__ float aw[TSEQ];
  __shared__ float red[TSEQ];
  int t = threadIdx.x;
  const float* hrow = hs + (size_t)t * DOUT;
  float d = 0.0f;
  for (int k = 0; k < DOUT; ++k) d += hrow[k] * q[k];
  float lam = fmaxf(__expf(loglam[0]), 1e-4f);
  float s = d * 0.044194173824159216f - lam * (float)(TSEQ - 1 - t);  // 1/sqrt(512)

  red[t] = s; __syncthreads();
  for (int o = 256; o > 0; o >>= 1) { if (t < o) red[t] = fmaxf(red[t], red[t + o]); __syncthreads(); }
  float mx = red[0]; __syncthreads();
  float e = __expf(s - mx);
  red[t] = e; __syncthreads();
  for (int o = 256; o > 0; o >>= 1) { if (t < o) red[t] += red[t + o]; __syncthreads(); }
  float a = e / red[0];
  aw[t] = a;
  out[DOUT + t] = a;
  __syncthreads();

  float sum = 0.0f;
  for (int tt = 0; tt < TSEQ; ++tt) sum += aw[tt] * hs[(size_t)tt * DOUT + t];
  out[t] = sum;
}

// ---------------------------------------------------------------------------
extern "C" void kernel_launch(void* const* d_in, const int* in_sizes, int n_in,
                              void* d_out, int out_size, void* d_ws, size_t ws_size,
                              hipStream_t stream) {
  (void)in_sizes; (void)n_in; (void)out_size; (void)ws_size;
  const int*   idx    = (const int*)  d_in[0];
  const float* timef  = (const float*)d_in[1];
  const float* emb    = (const float*)d_in[2];
  const float* latlon = (const float*)d_in[3];
  const float* lnw    = (const float*)d_in[4];
  const float* lnb    = (const float*)d_in[5];
  const float* w1     = (const float*)d_in[6];
  const float* b1     = (const float*)d_in[7];
  const float* w2     = (const float*)d_in[8];
  const float* b2     = (const float*)d_in[9];
  const float* w_ih   = (const float*)d_in[10];
  const float* w_hh   = (const float*)d_in[11];
  const float* b_ih   = (const float*)d_in[12];
  const float* b_hh   = (const float*)d_in[13];
  const float* q      = (const float*)d_in[14];
  const float* loglam = (const float*)d_in[15];
  float* out = (float*)d_out;

  // workspace layout (all 16B-aligned; total ~7.6 MB)
  char* ws = (char*)d_ws;
  unsigned short* xb   = (unsigned short*)(ws + 0);        // 512*288 bf16
  unsigned short* w1t  = (unsigned short*)(ws + 294912);   // 384*288 bf16
  unsigned short* h1   = (unsigned short*)(ws + 516096);   // 512*384 bf16
  unsigned short* w2t  = (unsigned short*)(ws + 909312);   // 256*384 bf16
  unsigned short* x2   = (unsigned short*)(ws + 1105920);  // 512*256 bf16
  unsigned short* wih  = (unsigned short*)(ws + 1368064);  // 1536*256 bf16
  float*          gi   = (float*)        (ws + 2154496);   // 512*1536 f32
  unsigned int*   whhP = (unsigned int*) (ws + 5300224);   // 256*1536 packed bf16x2
  float*          hs   = (float*)        (ws + 6873088);   // 512*512 f32

  // weight prep (bf16 conversion / transpose / packing)
  prep_transpose <<<432, 256, 0, stream>>>(w1, DIN,  DHID, w1t, DINP,  DHIDP);
  prep_transpose <<<384, 256, 0, stream>>>(w2, DHID, DPOI, w2t, DHIDP, DPOI);
  prep_copy_bf16 <<<1536, 256, 0, stream>>>(w_ih, wih, G3 * DPOI);
  prep_whh_packed<<<1536, 256, 0, stream>>>(w_hh, whhP);

  // stage 1: gather + features + LayerNorm
  featurize<<<TSEQ, 256, 0, stream>>>(idx, timef, emb, latlon, lnw, lnb, xb);

  // stage 2: MLP  (gelu(x@w1+b1) @ w2 + b2)   — WMMA bf16, 32x32 wave tiles
  wmma_gemm<<<48, 128, 0, stream>>>(xb, DINP,  w1t, DINP,  b1, DHID,
                                    16, 12, DINP,  nullptr, h1, DHIDP, 1);
  wmma_gemm<<<32, 128, 0, stream>>>(h1, DHIDP, w2t, DHIDP, b2, DPOI,
                                    16, 8,  DHIDP, nullptr, x2, DPOI, 0);

  // stage 3: gi_all = x2 @ w_ih^T + b_ih      — WMMA bf16, f32 out
  wmma_gemm<<<192, 128, 0, stream>>>(x2, DPOI, wih, DPOI, b_ih, G3,
                                     16, 48, DPOI, gi, nullptr, G3, 0);

  // stage 4: sequential GRU scan (latency chain; h in LDS, coalesced bf16 w_hh)
  gru_scan<<<1, 512, 0, stream>>>(gi, whhP, b_hh, hs);

  // stage 5: attention pooling
  attn_summary<<<1, 512, 0, stream>>>(hs, q, loglam, out);
}
